// TargetAwareContextAttention_91311004713262
// MI455X (gfx1250) — matrix operand order repaired and checked by reading, not compiled, not run on hardware
//
#include <hip/hip_runtime.h>
#include <hip/hip_bf16.h>
#include <math.h>

// ---------------- problem constants ----------------
#define BB    2
#define NT    128
#define NC    512
#define DD    256
#define DPHI  16
#define HID   128
#define NH    8
#define DK    32

typedef _Float16 f16;
typedef __attribute__((ext_vector_type(16))) _Float16 v16h;
typedef __attribute__((ext_vector_type(8)))  _Float16 v8h;
typedef __attribute__((ext_vector_type(8)))  float    v8f;
typedef __attribute__((ext_vector_type(4)))  unsigned int u32x4;
typedef __attribute__((ext_vector_type(8)))  unsigned int u32x8;

// ---------------- WMMA helpers ----------------
__device__ __forceinline__ v8f wmma_f16(v16h a, v16h b, v8f c) {
    // D(16x16,f32) = A(16x32,f16) x B(32x16,f16) + C
    return __builtin_amdgcn_wmma_f32_16x16x32_f16(
        /*neg_a=*/false, a, /*neg_b=*/false, b,
        /*c_mod=*/(short)0, c, /*reuse_a=*/false, /*reuse_b=*/false);
}

// A-matrix fragment from an LDS f16 array (row-major, leading dim ld),
// rows = tile M (ctx index), K-chunk starting at k0 (multiple of 32).
// ISA layout (16-bit A 16x32): lanes 0-15 hold M=lane, halves 0..7 -> K 0..7,
// halves 8..15 -> K 16..23; lanes 16-31 shift K by +8.
__device__ __forceinline__ v16h load_a_lds(const f16* base, int ld, int k0, int lane) {
    int M   = lane & 15;
    int hi8 = (lane >> 4) * 8;
    const f16* row = base + M * ld + k0 + hi8;
    union { v16h v; v8h h[2]; } u;
    u.h[0] = *(const v8h*)(row);        // K = k0+hi8   .. +7
    u.h[1] = *(const v8h*)(row + 16);   // K = k0+16+hi8 .. +7
    return u.v;
}

// B-matrix fragment from a pre-packed global f16 buffer.
// Packed layout: frag(nt,kt) = 32 lanes x 16 halves contiguous.
__device__ __forceinline__ v16h load_b_packed(const f16* pk, int ktiles, int nt, int kt, int lane) {
    return *(const v16h*)(pk + (((size_t)nt * ktiles + kt) * 32 + lane) * 16);
}

// ---------------- Tensor Data Mover: async 2-D tile -> LDS ----------------
// Builds a D# per CDNA5 ISA ch.8: group0 = {count/flags, lds_addr,
// global_addr[31:0], global_addr[56:32] | type=2}, group1 = {data_size=4B,
// tensor_dim0/1, tile_dim0/1, tensor_dim0_stride}. 2-D mode (VADDR2/3 = NULL).
// Wave-level instruction (EXEC ignored); issue from one wave only.
// Tracked by TENSORcnt; consumer gates on s_wait_tensorcnt.
__device__ __forceinline__ void tdm_load_2d_f32(unsigned int lds_off,
                                                const float* gptr,
                                                int rows, int cols, int row_stride) {
    unsigned long long ga = (unsigned long long)(uintptr_t)gptr;
    u32x4 g0;
    g0[0] = 0x1u;                                            // count=1, user mode
    g0[1] = lds_off;                                         // lds_addr (bytes)
    g0[2] = (unsigned int)ga;                                // global_addr[31:0]
    g0[3] = ((unsigned int)(ga >> 32) & 0x01FFFFFFu)         // global_addr[56:32]
            | (2u << 30);                                    // type = 2 (image)
    u32x8 g1;
    g1[0] = 2u << 16;                                        // data_size = 4 bytes
    g1[1] = ((unsigned int)cols & 0xFFFFu) << 16;            // tensor_dim0[15:0]
    g1[2] = (((unsigned int)cols >> 16) & 0xFFFFu)           // tensor_dim0[31:16]
            | (((unsigned int)rows & 0xFFFFu) << 16);        // tensor_dim1[15:0]
    g1[3] = (((unsigned int)rows >> 16) & 0xFFFFu)           // tensor_dim1[31:16]
            | (((unsigned int)cols & 0xFFFFu) << 16);        // tile_dim0
    g1[4] = (unsigned int)rows & 0xFFFFu;                    // tile_dim1 (tile_dim2=0)
    g1[5] = (unsigned int)row_stride;                        // tensor_dim0_stride[31:0]
    g1[6] = 0u;                                              // stride hi, dim1_stride lo
    g1[7] = 0u;
    asm volatile("tensor_load_to_lds %0, %1" :: "s"(g0), "s"(g1) : "memory");
}

// ---------------- prep kernels ----------------

// y[r][n] = dot(x[r][:K], w[n][:K]) + bias[n]  (torch Linear: y = x @ W^T + b)
__global__ void row_proj_kernel(const float* __restrict__ x,
                                const float* __restrict__ w,
                                const float* __restrict__ bias,
                                float* __restrict__ y, int K, int N) {
    int r = blockIdx.x;
    int n = threadIdx.x;           // blockDim.x == N
    float acc = bias ? bias[n] : 0.0f;
    const float* xr = x + (size_t)r * K;
    const float* wr = w + (size_t)n * K;
    for (int k = 0; k < K; ++k) acc = fmaf(xr[k], wr[k], acc);
    y[(size_t)r * N + n] = acc;
}

// Pack B = W^T (K x N) into WMMA-fragment order, f16.
// Fragment element (nt,kt,lane,h):  B[k][n] with
//   n = nt*16 + (lane&15),  k = kt*32 + (h>=8 ? 16 : 0) + (lane>=16 ? 8 : 0) + (h&7)
// Source: w[n][colOff + k] with row stride rowStride; k >= Ksrc pads with 0.
__global__ void pack_b_kernel(const float* __restrict__ w,
                              f16* __restrict__ dst,
                              int N, int Ksrc, int Kpad, int rowStride, int colOff) {
    int ktiles = Kpad / 32;
    int total  = (N / 16) * ktiles * 32 * 16;
    int idx = blockIdx.x * blockDim.x + threadIdx.x;
    if (idx >= total) return;
    int h    = idx & 15;
    int lane = (idx >> 4) & 31;
    int rest = idx >> 9;
    int kt   = rest % ktiles;
    int nt   = rest / ktiles;
    int n = nt * 16 + (lane & 15);
    int k = kt * 32 + ((h >> 3) << 4) + ((lane >> 4) << 3) + (h & 7);
    float v = (k < Ksrc) ? w[(size_t)n * rowStride + colOff + k] : 0.0f;
    dst[idx] = (f16)v;
}

// ---------------- main fused attention kernel ----------------
// One workgroup per (b, t). 128 threads = 4 wave32s.
__global__ __launch_bounds__(128)
void attn_kernel(const float* __restrict__ phi_t, const float* __restrict__ phi_c,
                 const unsigned char* __restrict__ mask,
                 const float* __restrict__ kcRc, const float* __restrict__ vcRc,
                 const float* __restrict__ ktRt, const float* __restrict__ vtRt,
                 const float* __restrict__ Qb,
                 const f16* __restrict__ kp1h, const f16* __restrict__ kp2h,
                 const f16* __restrict__ vp1h, const f16* __restrict__ vp2h,
                 const f16* __restrict__ g1h,  const f16* __restrict__ g2h,
                 const f16* __restrict__ g3h,
                 const float* __restrict__ kp1_b, const float* __restrict__ kp2_b,
                 const float* __restrict__ vp1_b, const float* __restrict__ vp2_b,
                 const float* __restrict__ g_b,
                 float* __restrict__ ctxb) {
    __shared__ __align__(16) float sQ[DD];
    __shared__ __align__(16) float sKt[DD];
    __shared__ __align__(16) float sVt[DD];
    __shared__ __align__(16) f16   sDphi[16][32];
    __shared__ __align__(16) f16   sHk[16][HID];
    __shared__ __align__(16) f16   sHv[16][HID];
    __shared__ __align__(16) float sKf[16][DD];     // K fp32, then Kg in place
    __shared__ __align__(16) float sVf[16][DD];     // V fp32, then Vg in place
    __shared__ __align__(16) f16   sKh[16][DD];
    __shared__ __align__(16) f16   sVh[16][DD];
    __shared__ __align__(16) f16   sAD[16][DD];
    __shared__ __align__(16) float sS[16][NH];
    // TDM double buffers for per-context row projections (16x256 f32 tiles)
    __shared__ __align__(16) float sKc[2][16][DD];
    __shared__ __align__(16) float sVc[2][16][DD];

    const int tid  = threadIdx.x;
    const int lane = tid & 31;
    const int wave = tid >> 5;
    const int b = blockIdx.x / NT;
    const int t = blockIdx.x % NT;
    const int row = b * NT + t;

    for (int i = tid; i < DD; i += 128) {
        sQ[i]  = Qb[(size_t)row * DD + i];
        sKt[i] = ktRt[(size_t)row * DD + i];
        sVt[i] = vtRt[(size_t)row * DD + i];
    }

    // Prologue: kick off async TDM loads of ctx tile 0 into buffer 0.
    if (wave == 0) {
        tdm_load_2d_f32((unsigned int)(uintptr_t)&sKc[0][0][0],
                        kcRc + (size_t)(b * NC) * DD, 16, DD, DD);
        tdm_load_2d_f32((unsigned int)(uintptr_t)&sVc[0][0][0],
                        vcRc + (size_t)(b * NC) * DD, 16, DD, DD);
    }

    // Online-softmax state: thread owns two (head,dim) accumulator slots.
    float acc[2]  = {0.0f, 0.0f};
    float mrun[2] = {-__builtin_inff(), -__builtin_inff()};
    float lrun[2] = {0.0f, 0.0f};
    const int h0 = tid >> 5;          // slot0 = tid        -> head 0..3
    const int d0 = tid & 31;
    const int h1 = 4 + (tid >> 5);    // slot1 = tid + 128  -> head 4..7
    const int d1 = tid & 31;

    const float* pt = phi_t + (size_t)row * DPHI;
    __syncthreads();

    for (int c0 = 0, it = 0; c0 < NC; c0 += 16, ++it) {
        const int par = it & 1;
        const bool has_next = (c0 + 16) < NC;

        // Prefetch next ctx tile into the other buffer while we compute.
        // (End-of-iteration barrier guarantees prior reads of par^1 are done.)
        if (wave == 0 && has_next) {
            tdm_load_2d_f32((unsigned int)(uintptr_t)&sKc[par ^ 1][0][0],
                            kcRc + (size_t)(b * NC + c0 + 16) * DD, 16, DD, DD);
            tdm_load_2d_f32((unsigned int)(uintptr_t)&sVc[par ^ 1][0][0],
                            vcRc + (size_t)(b * NC + c0 + 16) * DD, 16, DD, DD);
        }

        // ---- stage 1: dphi tile (16 ctx x 32, K-padded) in f16 ----
        for (int e = tid; e < 16 * 32; e += 128) {
            int i = e >> 5, k = e & 31;
            float v = (k < DPHI) ? (pt[k] - phi_c[((size_t)(b * NC + c0 + i)) * DPHI + k]) : 0.0f;
            sDphi[i][k] = (f16)v;
        }
        __syncthreads();

        // ---- stage 2: MLP layer 1 (16x32 @ 32x128), relu, to LDS f16 ----
        {
            v16h aD = load_a_lds(&sDphi[0][0], 32, 0, lane);
            for (int j = 0; j < 2; ++j) {
                int nt = wave * 2 + j;               // HID/16 = 8 tiles
                int Ncol = nt * 16 + (lane & 15);
                int mb = (lane >> 4) * 8;
                v8f ck = {0.f,0.f,0.f,0.f,0.f,0.f,0.f,0.f};
                ck = wmma_f16(aD, load_b_packed(kp1h, 1, nt, 0, lane), ck);
                float bk = kp1_b[Ncol];
                for (int r = 0; r < 8; ++r)
                    sHk[mb + r][Ncol] = (f16)fmaxf(ck[r] + bk, 0.0f);
                v8f cv = {0.f,0.f,0.f,0.f,0.f,0.f,0.f,0.f};
                cv = wmma_f16(aD, load_b_packed(vp1h, 1, nt, 0, lane), cv);
                float bv = vp1_b[Ncol];
                for (int r = 0; r < 8; ++r)
                    sHv[mb + r][Ncol] = (f16)fmaxf(cv[r] + bv, 0.0f);
            }
        }
        // Current tile's TDM loads must have landed before stage 3 consumes
        // them. Wave 0 gates on TENSORcnt (<=2 keeps only the prefetch of the
        // next tile outstanding); the barrier publishes to the other waves.
        if (wave == 0) {
            if (has_next) __builtin_amdgcn_s_wait_tensorcnt(2);
            else          __builtin_amdgcn_s_wait_tensorcnt(0);
        }
        __syncthreads();

        // ---- stage 3: MLP layer 2 (16x128 @ 128x256) + assemble K,V ----
        for (int j = 0; j < 4; ++j) {
            int nt = wave * 4 + j;                   // DD/16 = 16 tiles
            v8f ck = {0.f,0.f,0.f,0.f,0.f,0.f,0.f,0.f};
            v8f cv = {0.f,0.f,0.f,0.f,0.f,0.f,0.f,0.f};
            for (int kt = 0; kt < HID / 32; ++kt) {
                ck = wmma_f16(load_a_lds(&sHk[0][0], HID, kt * 32, lane),
                              load_b_packed(kp2h, HID / 32, nt, kt, lane), ck);
                cv = wmma_f16(load_a_lds(&sHv[0][0], HID, kt * 32, lane),
                              load_b_packed(vp2h, HID / 32, nt, kt, lane), cv);
            }
            int Ncol = nt * 16 + (lane & 15);
            int mb = (lane >> 4) * 8;
            float bk = kp2_b[Ncol], bv = vp2_b[Ncol];
            float kt_n = sKt[Ncol], vt_n = sVt[Ncol];
            for (int r = 0; r < 8; ++r) {
                int M = mb + r;
                float Kv = ck[r] + bk + sKc[par][M][Ncol] + kt_n;
                float Vv = cv[r] + bv + sVc[par][M][Ncol] + vt_n;
                sKf[M][Ncol] = Kv;
                sVf[M][Ncol] = Vv;
                sKh[M][Ncol] = (f16)Kv;
                sVh[M][Ncol] = (f16)Vv;
                sAD[M][Ncol] = (f16)fabsf(Kv - Vv);
            }
        }
        __syncthreads();

        // ---- stage 4: gate = sigmoid(K@gw1^T + V@gw2^T + |K-V|@gw3^T + b) ----
        for (int j = 0; j < 4; ++j) {
            int nt = wave * 4 + j;
            v8f cg = {0.f,0.f,0.f,0.f,0.f,0.f,0.f,0.f};
            for (int kt = 0; kt < DD / 32; ++kt) {
                cg = wmma_f16(load_a_lds(&sKh[0][0], DD, kt * 32, lane),
                              load_b_packed(g1h, DD / 32, nt, kt, lane), cg);
                cg = wmma_f16(load_a_lds(&sVh[0][0], DD, kt * 32, lane),
                              load_b_packed(g2h, DD / 32, nt, kt, lane), cg);
                cg = wmma_f16(load_a_lds(&sAD[0][0], DD, kt * 32, lane),
                              load_b_packed(g3h, DD / 32, nt, kt, lane), cg);
            }
            int Ncol = nt * 16 + (lane & 15);
            int mb = (lane >> 4) * 8;
            float bg = g_b[Ncol];
            for (int r = 0; r < 8; ++r) {
                int M = mb + r;
                float g = 1.0f / (1.0f + __expf(-(cg[r] + bg)));
                sKf[M][Ncol] *= g;     // Kg in place
                sVf[M][Ncol] *= g;     // Vg in place
            }
        }
        __syncthreads();

        // ---- stage 5: scores (16 ctx x 8 heads == 128 threads) ----
        {
            int i  = tid >> 3;
            int hh = tid & 7;
            const float* q  = &sQ[hh * DK];
            const float* kg = &sKf[i][hh * DK];
            float s = 0.0f;
            for (int k = 0; k < DK; ++k) s = fmaf(q[k], kg[k], s);
            s *= 0.17677669529663687f;             // 1/sqrt(32)
            if (!mask[b * NC + c0 + i]) s = -__builtin_inff();
            sS[i][hh] = s;
        }
        __syncthreads();

        // ---- stage 6: online softmax accumulate (2 slots per thread) ----
        for (int sl = 0; sl < 2; ++sl) {
            int hh = sl ? h1 : h0;
            int dd = sl ? d1 : d0;
            float mnew = mrun[sl];
            for (int i = 0; i < 16; ++i) mnew = fmaxf(mnew, sS[i][hh]);
            if (mnew != -__builtin_inff()) {
                float resc = __expf(mrun[sl] - mnew);   // exp(-inf)=0 when fresh
                float a = acc[sl] * resc;
                float l = lrun[sl] * resc;
                for (int i = 0; i < 16; ++i) {
                    float p = __expf(sS[i][hh] - mnew); // masked -> exp(-inf)=0
                    a = fmaf(p, sVf[i][hh * DK + dd], a);
                    l += p;
                }
                acc[sl] = a; lrun[sl] = l; mrun[sl] = mnew;
            }
        }
        __syncthreads();
    }

    ctxb[(size_t)row * DD + tid]       = acc[0] / lrun[0];
    ctxb[(size_t)row * DD + tid + 128] = acc[1] / lrun[1];
}

// out[r][n] = dot(ctx[r], out_w[n]) + out_b[n]
__global__ void out_proj_kernel(const float* __restrict__ ctxb,
                                const float* __restrict__ w,
                                const float* __restrict__ bias,
                                float* __restrict__ out) {
    int r = blockIdx.x;
    int n = threadIdx.x;   // 256
    float a = bias[n];
    const float* xr = ctxb + (size_t)r * DD;
    const float* wr = w + (size_t)n * DD;
    for (int k = 0; k < DD; ++k) a = fmaf(xr[k], wr[k], a);
    out[(size_t)r * DD + n] = a;
}

// ---------------- launcher ----------------
static inline size_t align_up(size_t x, size_t a) { return (x + a - 1) & ~(a - 1); }

extern "C" void kernel_launch(void* const* d_in, const int* in_sizes, int n_in,
                              void* d_out, int out_size, void* d_ws, size_t ws_size,
                              hipStream_t stream) {
    const float* R_t    = (const float*)d_in[0];
    const float* R_ctx  = (const float*)d_in[1];
    const float* phi_t  = (const float*)d_in[2];
    const float* phi_c  = (const float*)d_in[3];
    const unsigned char* mask = (const unsigned char*)d_in[4];
    const float* Wq_w  = (const float*)d_in[5];
    const float* Wq_b  = (const float*)d_in[6];
    const float* kc_w  = (const float*)d_in[7];
    const float* kt_w  = (const float*)d_in[8];
    const float* kp1_w = (const float*)d_in[9];
    const float* kp1_b = (const float*)d_in[10];
    const float* kp2_w = (const float*)d_in[11];
    const float* kp2_b = (const float*)d_in[12];
    const float* vc_w  = (const float*)d_in[13];
    const float* vt_w  = (const float*)d_in[14];
    const float* vp1_w = (const float*)d_in[15];
    const float* vp1_b = (const float*)d_in[16];
    const float* vp2_w = (const float*)d_in[17];
    const float* vp2_b = (const float*)d_in[18];
    const float* g_w   = (const float*)d_in[19];
    const float* g_b   = (const float*)d_in[20];
    const float* out_w = (const float*)d_in[21];
    const float* out_b = (const float*)d_in[22];
    float* outp = (float*)d_out;

    // ---- workspace layout ----
    char* ws = (char*)d_ws;
    size_t off = 0;
    auto take = [&](size_t bytes) { size_t o = off; off = align_up(off + bytes, 256); return o; };
    float* kcRc = (float*)(ws + take((size_t)BB * NC * DD * 4));
    float* vcRc = (float*)(ws + take((size_t)BB * NC * DD * 4));
    float* ktRt = (float*)(ws + take((size_t)BB * NT * DD * 4));
    float* vtRt = (float*)(ws + take((size_t)BB * NT * DD * 4));
    float* Qb   = (float*)(ws + take((size_t)BB * NT * DD * 4));
    float* ctxb = (float*)(ws + take((size_t)BB * NT * DD * 4));
    f16* kp1h = (f16*)(ws + take((size_t)(HID/16) * (32/32)  * 512 * 2));
    f16* kp2h = (f16*)(ws + take((size_t)(DD/16)  * (HID/32) * 512 * 2));
    f16* vp1h = (f16*)(ws + take((size_t)(HID/16) * (32/32)  * 512 * 2));
    f16* vp2h = (f16*)(ws + take((size_t)(DD/16)  * (HID/32) * 512 * 2));
    f16* g1h  = (f16*)(ws + take((size_t)(DD/16)  * (DD/32)  * 512 * 2));
    f16* g2h  = (f16*)(ws + take((size_t)(DD/16)  * (DD/32)  * 512 * 2));
    f16* g3h  = (f16*)(ws + take((size_t)(DD/16)  * (DD/32)  * 512 * 2));

    // ---- prep: row projections (fp32, small; L2-resident) ----
    row_proj_kernel<<<BB * NC, DD, 0, stream>>>(R_ctx, kc_w, nullptr, kcRc, DD, DD);
    row_proj_kernel<<<BB * NC, DD, 0, stream>>>(R_ctx, vc_w, nullptr, vcRc, DD, DD);
    row_proj_kernel<<<BB * NT, DD, 0, stream>>>(R_t,   kt_w, nullptr, ktRt, DD, DD);
    row_proj_kernel<<<BB * NT, DD, 0, stream>>>(R_t,   vt_w, nullptr, vtRt, DD, DD);
    row_proj_kernel<<<BB * NT, DD, 0, stream>>>(R_t,   Wq_w, Wq_b,    Qb,   DD, DD);

    // ---- prep: pack weights into f16 WMMA-fragment order ----
    // (N, Ksrc, Kpad, rowStride, colOff)
    {
        int tot;
        tot = (HID/16) * 1 * 512;
        pack_b_kernel<<<(tot + 255) / 256, 256, 0, stream>>>(kp1_w, kp1h, HID, DPHI, 32, DPHI, 0);
        pack_b_kernel<<<(tot + 255) / 256, 256, 0, stream>>>(vp1_w, vp1h, HID, DPHI, 32, DPHI, 0);
        tot = (DD/16) * (HID/32) * 512;
        pack_b_kernel<<<(tot + 255) / 256, 256, 0, stream>>>(kp2_w, kp2h, DD, HID, HID, HID, 0);
        pack_b_kernel<<<(tot + 255) / 256, 256, 0, stream>>>(vp2_w, vp2h, DD, HID, HID, HID, 0);
        tot = (DD/16) * (DD/32) * 512;
        pack_b_kernel<<<(tot + 255) / 256, 256, 0, stream>>>(g_w, g1h, DD, DD, DD, 3 * DD, 0);
        pack_b_kernel<<<(tot + 255) / 256, 256, 0, stream>>>(g_w, g2h, DD, DD, DD, 3 * DD, DD);
        pack_b_kernel<<<(tot + 255) / 256, 256, 0, stream>>>(g_w, g3h, DD, DD, DD, 3 * DD, 2 * DD);
    }

    // ---- main fused kernel: 256 targets x 4 waves, WMMA + TDM prefetch ----
    attn_kernel<<<BB * NT, 128, 0, stream>>>(
        phi_t, phi_c, mask,
        kcRc, vcRc, ktRt, vtRt, Qb,
        kp1h, kp2h, vp1h, vp2h, g1h, g2h, g3h,
        kp1_b, kp2_b, vp1_b, vp2_b, g_b,
        ctxb);

    // ---- epilogue: out projection ----
    out_proj_kernel<<<BB * NT, DD, 0, stream>>>(ctxb, out_w, out_b, outp);
}
